// AttentionSHDSNN_15831249453644
// MI455X (gfx1250) — matrix operand order repaired and checked by reading, not compiled, not run on hardware
//
#include <hip/hip_runtime.h>

// ---------------------------------------------------------------------------
// Spiking attention SNN forward, CDNA5 (gfx1250), wave32 WMMA f16.
// One workgroup per 16-row batch tile runs all 250 timesteps (batch rows are
// independent through the recurrence). 1024 threads = 32 waves per WG; each
// wave owns ONE 16x16 N-tile (16 cols) of every E=512-wide GEMM, keeping the
// recurrent state v1/a1 in WMMA C-fragment registers with low VGPR pressure
// (no scratch spills). s1 / v1n / av / h staged in LDS (f16) as WMMA A
// operands. Weights converted once to f16 in d_ws and streamed from L2
// directly in B-fragment layout. The K=700 boundary tile of the input GEMM is
// handled branchlessly (clamped-address vector load + 0/1 mask) and runs
// FIRST so it initializes the accumulator and does not overlap the pipelined
// main loop.
// ---------------------------------------------------------------------------

typedef __attribute__((ext_vector_type(16))) _Float16 v16h;
typedef __attribute__((ext_vector_type(8)))  float    v8f;

#define T_STEPS 250
#define N_IN    700
#define N_INP   704       // K padded to multiple of 32
#define E_DIM   512
#define N_HEAD  8
#define N_OUT   20
#define BTILE   16
#define NTHR    1024      // 32 waves

union HU { v16h v; uint4 q[2]; };

__device__ __forceinline__ v8f wmma16(v16h a, v16h b, v8f c) {
  // D = A(16x32 f16) * B(32x16 f16) + C(16x16 f32)
  return __builtin_amdgcn_wmma_f32_16x16x32_f16(
      /*neg_a=*/false, a, /*neg_b=*/false, b,
      /*c_mod=*/(short)0, c, /*reuse_a=*/false, /*reuse_b=*/false);
}

// B fragment (32x16, KxN): lane&15 = column; lanes 0-15 hold K=kt..+7 & kt+16..+23,
// lanes 16-31 hold K=kt+8..+15 & kt+24..+31 (matches A fragment order).
__device__ __forceinline__ v16h load_b_frag(const _Float16* __restrict__ W,
                                            int stride, int n0, int kt) {
  const int lane = threadIdx.x & 31;
  const _Float16* p = W + (size_t)(n0 + (lane & 15)) * stride
                        + (kt + ((lane < 16) ? 0 : 8));
  HU u;
  u.q[0] = *(const uint4*)(p);
  u.q[1] = *(const uint4*)(p + 16);
  return u.v;
}

// A fragment (16x32, MxK) from LDS row-major [16][E_DIM]:
// lane&15 = row; lanes 0-15: halves 0-7 -> K=kt..+7, halves 8-15 -> K=kt+16..+23
// lanes 16-31: +8 on both K groups.
__device__ __forceinline__ v16h load_a_lds(const _Float16* A, int kt) {
  const int lane = threadIdx.x & 31;
  const _Float16* p = A + (lane & 15) * E_DIM + (kt + ((lane < 16) ? 0 : 8));
  HU u;
  u.q[0] = *(const uint4*)(p);
  u.q[1] = *(const uint4*)(p + 16);
  return u.v;
}

// A fragment for x (f32 in global), full in-bounds tile.
__device__ __forceinline__ v16h load_a_x(const float* __restrict__ px, int ks) {
  float4 f0 = *(const float4*)(px + ks);
  float4 f1 = *(const float4*)(px + ks + 4);
  float4 f2 = *(const float4*)(px + ks + 16);
  float4 f3 = *(const float4*)(px + ks + 20);
  v16h a;
  a[0]=(_Float16)f0.x;  a[1]=(_Float16)f0.y;  a[2]=(_Float16)f0.z;  a[3]=(_Float16)f0.w;
  a[4]=(_Float16)f1.x;  a[5]=(_Float16)f1.y;  a[6]=(_Float16)f1.z;  a[7]=(_Float16)f1.w;
  a[8]=(_Float16)f2.x;  a[9]=(_Float16)f2.y;  a[10]=(_Float16)f2.z; a[11]=(_Float16)f2.w;
  a[12]=(_Float16)f3.x; a[13]=(_Float16)f3.y; a[14]=(_Float16)f3.z; a[15]=(_Float16)f3.w;
  return a;
}

// Boundary tile (kt=672), branchless. ks = 672 (lanes 0-15) or 680 (lanes
// 16-31). Lanes 0-15 cover K 672..679 & 688..695 (all valid). Lanes 16-31
// cover K 680..687 & 696..703: only the final float4 (700..703) is OOB ->
// clamp its address to a valid offset and zero the result with a 0/1 mask.
__device__ __forceinline__ v16h load_a_x_tail(const float* __restrict__ px, int ks) {
  float4 f0 = *(const float4*)(px + ks);
  float4 f1 = *(const float4*)(px + ks + 4);
  float4 f2 = *(const float4*)(px + ks + 16);
  const bool hi_ok = (ks + 23) < N_IN;                 // false only for lanes>=16
  float4 f3 = *(const float4*)(px + (hi_ok ? ks + 20 : ks));
  const float m = hi_ok ? 1.0f : 0.0f;
  v16h a;
  a[0]=(_Float16)f0.x;      a[1]=(_Float16)f0.y;      a[2]=(_Float16)f0.z;      a[3]=(_Float16)f0.w;
  a[4]=(_Float16)f1.x;      a[5]=(_Float16)f1.y;      a[6]=(_Float16)f1.z;      a[7]=(_Float16)f1.w;
  a[8]=(_Float16)f2.x;      a[9]=(_Float16)f2.y;      a[10]=(_Float16)f2.z;     a[11]=(_Float16)f2.w;
  a[12]=(_Float16)(f3.x*m); a[13]=(_Float16)(f3.y*m); a[14]=(_Float16)(f3.z*m); a[15]=(_Float16)(f3.w*m);
  return a;
}

__global__ void convert_pad(const float* __restrict__ src, _Float16* __restrict__ dst,
                            int srcRows, int srcCols, int dstRows, int dstCols) {
  int idx = blockIdx.x * blockDim.x + threadIdx.x;
  int total = dstRows * dstCols;
  if (idx >= total) return;
  int r = idx / dstCols;
  int c = idx - r * dstCols;
  float v = (r < srcRows && c < srcCols) ? src[(size_t)r * srcCols + c] : 0.0f;
  dst[idx] = (_Float16)v;
}

__global__ __launch_bounds__(NTHR, 1)
void snn_kernel(const float* __restrict__ x,
                const _Float16* __restrict__ w1p,
                const _Float16* __restrict__ wrec,
                const _Float16* __restrict__ wq,
                const _Float16* __restrict__ wk,
                const _Float16* __restrict__ wvm,
                const _Float16* __restrict__ wo,
                const float* __restrict__ ln_g,
                const float* __restrict__ ln_b,
                const _Float16* __restrict__ wfo,
                const float* __restrict__ alpha,
                const float* __restrict__ rho,
                const float* __restrict__ beta_a,
                const float* __restrict__ beta_out,
                float* __restrict__ out) {
  __shared__ __align__(16) _Float16 sS1[BTILE * E_DIM];   // spikes (A operand)
  __shared__ __align__(16) _Float16 sV1H[BTILE * E_DIM];  // v1n, later h
  __shared__ __align__(16) _Float16 sAV[BTILE * E_DIM];   // attn-weighted V
  __shared__ float sScore[BTILE * N_HEAD];
  __shared__ float sAttn[BTILE * N_HEAD];
  __shared__ float sSum[BTILE];
  __shared__ float sSqs[BTILE];

  const int tid    = threadIdx.x;
  const int lane   = tid & 31;
  const int wv_id  = tid >> 5;             // 0..31
  const int n0     = wv_id * 16;           // wave's single N-tile base column
  const int lrow   = lane & 15;
  const int rowoff = (lane < 16) ? 0 : 8;  // C-fragment row offset
  const int koffA  = (lane < 16) ? 0 : 8;  // A-fragment K offset
  const int head   = wv_id >> 2;           // 16 cols fall in exactly one head
  const int b0     = blockIdx.x * BTILE;
  const int col    = n0 + lrow;
  const float scale = 0.125f;              // 1/sqrt(64)

  // per-lane column parameters
  const float al = alpha[col];
  const float rh = rho[col];
  const float ba = beta_a[col];
  const float g  = ln_g[col];
  const float lb = ln_b[col];

  // recurrent state in C-fragment layout (one N-tile)
  v8f v1s = {}, a1s = {};
  // readout state (waves 0,1 only; cols 0-15 / 16-31)
  v8f voutc = {}, accc = {};
  float bout_c = 0.0f;
  if (wv_id < 2 && col < N_OUT) bout_c = beta_out[col];

  for (int i = tid; i < BTILE * E_DIM; i += NTHR) sS1[i] = (_Float16)0.0f;
  __syncthreads();

  for (int t = 0; t < T_STEPS; ++t) {
    // ---------------- I1 = x_t @ W1^T + s1 @ Wrec^T ----------------
    const float* px = x + ((size_t)(b0 + lrow) * T_STEPS + t) * N_IN;
    // boundary tile first: initializes the accumulator (inline-0 C operand),
    // keeps its fragments out of the pipelined main loop's live ranges.
    v8f i1;
    {
      v16h a = load_a_x_tail(px, 672 + koffA);
      v8f cz = {};
      i1 = wmma16(a, load_b_frag(w1p, N_INP, n0, 672), cz);
    }
#pragma unroll 1
    for (int kt = 0; kt < 672; kt += 32) {
      v16h a = load_a_x(px, kt + koffA);
      i1 = wmma16(a, load_b_frag(w1p, N_INP, n0, kt), i1);
    }
#pragma unroll 2
    for (int kt = 0; kt < E_DIM; kt += 32) {
      v16h a = load_a_lds(sS1, kt);
      i1 = wmma16(a, load_b_frag(wrec, E_DIM, n0, kt), i1);
    }
    __syncthreads();  // all reads of old sS1 done

    // ---------------- adaptive LIF update (state in regs) ----------------
#pragma unroll
    for (int r = 0; r < 8; ++r) {
      int row = r + rowoff;
      float s1p  = (float)sS1[row * E_DIM + col];
      float v1n  = al * v1s[r] * (1.0f - s1p) + (1.0f - al) * (i1[r] - a1s[r]);
      float sp   = (v1n > 1.0f) ? 1.0f : 0.0f;
      a1s[r] = rh * a1s[r] + ba * sp;
      v1s[r] = v1n;
      sS1[row * E_DIM + col]  = (_Float16)sp;
      sV1H[row * E_DIM + col] = (_Float16)v1n;
    }
    if (tid < BTILE * N_HEAD) sScore[tid] = 0.0f;
    if (tid < BTILE) { sSum[tid] = 0.0f; sSqs[tid] = 0.0f; }
    __syncthreads();

    // ---------------- Q,K from v1n (separate passes, low pressure) -------
    v8f qa = {};
#pragma unroll 2
    for (int kt = 0; kt < E_DIM; kt += 32) {
      v16h a = load_a_lds(sV1H, kt);
      qa = wmma16(a, load_b_frag(wq, E_DIM, n0, kt), qa);
    }
    v8f ka = {};
#pragma unroll 2
    for (int kt = 0; kt < E_DIM; kt += 32) {
      v16h a = load_a_lds(sV1H, kt);
      ka = wmma16(a, load_b_frag(wk, E_DIM, n0, kt), ka);
    }
    // per-head score partials: q,k for same (row,col) sit in same lane/VGPR;
    // qa/ka die here.
#pragma unroll
    for (int r = 0; r < 8; ++r) {
      int row = r + rowoff;
      atomicAdd(&sScore[row * N_HEAD + head], qa[r] * ka[r]);
    }

    // ---------------- V from spikes ----------------
    v8f va = {};
#pragma unroll 2
    for (int kt = 0; kt < E_DIM; kt += 32) {
      v16h a = load_a_lds(sS1, kt);
      va = wmma16(a, load_b_frag(wvm, E_DIM, n0, kt), va);
    }
    __syncthreads();

    // ---------------- softmax over 8 heads ----------------
    if (tid < BTILE * N_HEAD) {
      int row = tid >> 3, h = tid & 7;
      float mx = -1e30f;
#pragma unroll
      for (int j = 0; j < N_HEAD; ++j)
        mx = fmaxf(mx, sScore[row * N_HEAD + j] * scale);
      float den = 0.0f;
#pragma unroll
      for (int j = 0; j < N_HEAD; ++j)
        den += __expf(sScore[row * N_HEAD + j] * scale - mx);
      sAttn[row * N_HEAD + h] =
          __expf(sScore[row * N_HEAD + h] * scale - mx) / den;
    }
    __syncthreads();

    // ---------------- attn-weighted V -> LDS (A operand) ----------------
#pragma unroll
    for (int r = 0; r < 8; ++r) {
      int row = r + rowoff;
      float at = sAttn[row * N_HEAD + head];
      sAV[row * E_DIM + col] = (_Float16)(at * va[r]);
    }
    __syncthreads();

    // ---------------- attn_out = av @ Wo^T ----------------
    v8f ao = {};
#pragma unroll 2
    for (int kt = 0; kt < E_DIM; kt += 32) {
      v16h a = load_a_lds(sAV, kt);
      ao = wmma16(a, load_b_frag(wo, E_DIM, n0, kt), ao);
    }

    // ---------------- y = s1 + attn_out; LayerNorm stats ----------------
    float y[8];
#pragma unroll
    for (int r = 0; r < 8; ++r) {
      int row = r + rowoff;
      y[r] = (float)sS1[row * E_DIM + col] + ao[r];
      atomicAdd(&sSum[row], y[r]);
      atomicAdd(&sSqs[row], y[r] * y[r]);
    }
    __syncthreads();

    // ---------------- h = LN(y)*g + b -> LDS (reuse sV1H) ----------------
#pragma unroll
    for (int r = 0; r < 8; ++r) {
      int row = r + rowoff;
      float mu  = sSum[row] * (1.0f / E_DIM);
      float var = sSqs[row] * (1.0f / E_DIM) - mu * mu;
      float rs  = rsqrtf(var + 1e-5f);
      sV1H[row * E_DIM + col] = (_Float16)((y[r] - mu) * rs * g + lb);
    }
    __syncthreads();

    // ---------------- readout: i_out = h @ Wfc^T (waves 0,1) -------------
    if (wv_id < 2) {
      v8f io = {};
#pragma unroll 2
      for (int kt = 0; kt < E_DIM; kt += 32) {
        v16h a = load_a_lds(sV1H, kt);
        io = wmma16(a, load_b_frag(wfo, E_DIM, n0, kt), io);
      }
#pragma unroll
      for (int r = 0; r < 8; ++r) {
        voutc[r] = bout_c * voutc[r] + io[r];
        accc[r] += voutc[r];
      }
    }
    __syncthreads();
  }

  // ---------------- write mean over time ----------------
  if (wv_id < 2 && col < N_OUT) {
#pragma unroll
    for (int r = 0; r < 8; ++r) {
      int row = r + rowoff;
      out[(size_t)(b0 + row) * N_OUT + col] = accc[r] * (1.0f / T_STEPS);
    }
  }
}

extern "C" void kernel_launch(void* const* d_in, const int* in_sizes, int n_in,
                              void* d_out, int out_size, void* d_ws, size_t ws_size,
                              hipStream_t stream) {
  const float* x        = (const float*)d_in[0];
  const float* W1       = (const float*)d_in[1];
  const float* Wrec     = (const float*)d_in[2];
  const float* Wq       = (const float*)d_in[3];
  const float* Wk       = (const float*)d_in[4];
  const float* Wv       = (const float*)d_in[5];
  const float* Wo       = (const float*)d_in[6];
  const float* ln_g     = (const float*)d_in[7];
  const float* ln_b     = (const float*)d_in[8];
  const float* Wfc      = (const float*)d_in[9];
  const float* alpha    = (const float*)d_in[10];
  const float* rho      = (const float*)d_in[11];
  const float* beta_a   = (const float*)d_in[12];
  const float* beta_out = (const float*)d_in[13];
  float* out = (float*)d_out;

  // f16 weight arena in workspace
  _Float16* w1p  = (_Float16*)d_ws;                       // [512][704]
  _Float16* wrec = w1p  + (size_t)E_DIM * N_INP;          // [512][512]
  _Float16* wq   = wrec + (size_t)E_DIM * E_DIM;
  _Float16* wk   = wq   + (size_t)E_DIM * E_DIM;
  _Float16* wv   = wk   + (size_t)E_DIM * E_DIM;
  _Float16* wo   = wv   + (size_t)E_DIM * E_DIM;
  _Float16* wfo  = wo   + (size_t)E_DIM * E_DIM;          // [32][512], rows >=20 zero

  const int B = in_sizes[0] / (T_STEPS * N_IN);

  auto nb = [](int n) { return (n + 255) / 256; };
  convert_pad<<<nb(E_DIM * N_INP), 256, 0, stream>>>(W1,   w1p,  E_DIM, N_IN,  E_DIM, N_INP);
  convert_pad<<<nb(E_DIM * E_DIM), 256, 0, stream>>>(Wrec, wrec, E_DIM, E_DIM, E_DIM, E_DIM);
  convert_pad<<<nb(E_DIM * E_DIM), 256, 0, stream>>>(Wq,   wq,   E_DIM, E_DIM, E_DIM, E_DIM);
  convert_pad<<<nb(E_DIM * E_DIM), 256, 0, stream>>>(Wk,   wk,   E_DIM, E_DIM, E_DIM, E_DIM);
  convert_pad<<<nb(E_DIM * E_DIM), 256, 0, stream>>>(Wv,   wv,   E_DIM, E_DIM, E_DIM, E_DIM);
  convert_pad<<<nb(E_DIM * E_DIM), 256, 0, stream>>>(Wo,   wo,   E_DIM, E_DIM, E_DIM, E_DIM);
  convert_pad<<<nb(32 * E_DIM),    256, 0, stream>>>(Wfc,  wfo,  N_OUT, E_DIM, 32,    E_DIM);

  snn_kernel<<<B / BTILE, NTHR, 0, stream>>>(x, w1p, wrec, wq, wk, wv, wo,
                                             ln_g, ln_b, wfo, alpha, rho,
                                             beta_a, beta_out, out);
}
